// VectorQuantizer_30288109371954
// MI455X (gfx1250) — matrix operand order repaired and checked by reading, not compiled, not run on hardware
//
#include <hip/hip_runtime.h>
#include <hip/hip_bf16.h>

typedef __attribute__((ext_vector_type(16))) __bf16 v16bf;
typedef __attribute__((ext_vector_type(8)))  float  v8f;

#define NUM_CODES 1024
#define DIM       256
#define HW        1024      // 32*32
#define CHW       (256*1024)
#define TOT_ELEMS 16777216  // 64*256*32*32

// ---------------- codebook norms ----------------
__global__ void k_cnorm(const float* __restrict__ cb, float* __restrict__ cnorm) {
    int k = blockIdx.x * blockDim.x + threadIdx.x;
    if (k >= NUM_CODES) return;
    const float* row = cb + (size_t)k * DIM;
    float s = 0.f;
    #pragma unroll 8
    for (int c = 0; c < DIM; ++c) s += row[c] * row[c];
    cnorm[k] = s;
}

// ---- pre-swizzle codebook into per-lane WMMA B-operand layout (bf16 hi/lo) ----
// B operand (K=32 x N=16), k order chosen here (same as A): lane n in 0..15 holds column n; lanes 16..31 hold
// same column with K offset +16; dwords pack K sequentially (2 bf16/dword).
__global__ void k_bswz(const float* __restrict__ cb, __bf16* __restrict__ bswz) {
    int tid  = blockIdx.x * blockDim.x + threadIdx.x;   // 16384 = 64ct * 8kc * 32lanes
    int ct   = tid >> 8;
    int kc   = (tid >> 5) & 7;
    int lane = tid & 31;
    int n    = lane & 15, half = lane >> 4;
    __bf16* dst = bswz + (size_t)tid * 32;              // 32 bf16 = 64B per lane
    const float* src = cb + (size_t)(ct * 16 + n) * DIM + kc * 32 + half * 16;
    #pragma unroll
    for (int i = 0; i < 16; ++i) {
        float f = src[i];
        __bf16 hi = (__bf16)f;
        __bf16 lo = (__bf16)(f - (float)hi);
        dst[i]      = hi;
        dst[16 + i] = lo;
    }
}

// ---------------- fused GEMM (split-bf16 WMMA) + row argmin ----------------
__global__ void __launch_bounds__(256) k_main(const float* __restrict__ x,
                                              const __bf16* __restrict__ bswz,
                                              const float* __restrict__ cnorm,
                                              int* __restrict__ idxout) {
    int lane = threadIdx.x & 31;
    int wave = threadIdx.x >> 5;
    int t    = blockIdx.x * 8 + wave;        // row tile id, 4096 total
    int m    = lane & 15, half = lane >> 4;
    int r    = t * 16 + m;                   // global pixel row
    int b    = r >> 10, hw = r & 1023;
    const float* xb = x + (size_t)b * CHW + hw;   // + k*HW walks channels

    // A operand 16x32 bf16, k order chosen here (same as B): lanes 0-15 = rows, K = {0..15};
    // lanes 16-31 same rows, K = {16..31}. Fold -2x into A.
    v16bf ahi[8], alo[8];
    #pragma unroll
    for (int kc = 0; kc < 8; ++kc) {
        #pragma unroll
        for (int i = 0; i < 16; ++i) {
            int k   = kc * 32 + half * 16 + i;
            float f = -2.0f * xb[(size_t)k * HW];
            __bf16 hi = (__bf16)f;
            ahi[kc][i] = hi;
            alo[kc][i] = (__bf16)(f - (float)hi);
        }
    }

    float vbest[8];
    int   ibest[8];
    #pragma unroll
    for (int g = 0; g < 8; ++g) { vbest[g] = 3.4e38f; ibest[g] = 0; }

    const v16bf* bp = (const v16bf*)bswz;
    for (int ct = 0; ct < 64; ++ct) {
        v8f acc = {};
        #pragma unroll
        for (int kc = 0; kc < 8; ++kc) {
            int base = ((ct * 8 + kc) * 32 + lane) * 2;
            v16bf bhi = bp[base];
            v16bf blo = bp[base + 1];
            // score += Xhi*Chi + Xhi*Clo + Xlo*Chi  (~fp32 accuracy)
            acc = __builtin_amdgcn_wmma_f32_16x16x32_bf16(false, ahi[kc], false, bhi,
                                                          (short)0, acc, false, false);
            acc = __builtin_amdgcn_wmma_f32_16x16x32_bf16(false, ahi[kc], false, blo,
                                                          (short)0, acc, false, false);
            acc = __builtin_amdgcn_wmma_f32_16x16x32_bf16(false, alo[kc], false, bhi,
                                                          (short)0, acc, false, false);
        }
        // D layout: lane holds column n = lane&15; VGPR g -> row g (+8 if lane>=16)
        int   code = ct * 16 + (lane & 15);
        float cn   = cnorm[code];
        #pragma unroll
        for (int g = 0; g < 8; ++g) {
            float cand = cn + acc[g];          // ||c||^2 - 2 x.c  (||x||^2 const)
            if (cand < vbest[g]) { vbest[g] = cand; ibest[g] = code; }
        }
    }

    // argmin across the 16 lanes of each half (wave32 shuffles), tiebreak low idx
    #pragma unroll
    for (int g = 0; g < 8; ++g) {
        float v  = vbest[g];
        int   ix = ibest[g];
        #pragma unroll
        for (int mask = 1; mask < 16; mask <<= 1) {
            float ov = __shfl_xor(v, mask, 32);
            int   oi = __shfl_xor(ix, mask, 32);
            if (ov < v || (ov == v && oi < ix)) { v = ov; ix = oi; }
        }
        ibest[g] = ix;
    }
    if ((lane & 15) == 0) {
        #pragma unroll
        for (int g = 0; g < 8; ++g)
            idxout[t * 16 + half * 8 + g] = ibest[g];
    }
}

// ---------------- gather + straight-through output + loss partials ----------------
__global__ void __launch_bounds__(256) k_gather(const float* __restrict__ x,
                                                const float* __restrict__ cb,
                                                const int* __restrict__ idx,
                                                float* __restrict__ out,
                                                float* __restrict__ partials) {
    __shared__ float sred[256];
    int e   = blockIdx.x * 256 + threadIdx.x;   // element in NCHW order
    int b   = e >> 18;                          // / 262144
    int rem = e & 262143;
    int c   = rem >> 10;
    int hw  = rem & 1023;
    int r   = (b << 10) | hw;                   // flattened pixel row
    float q  = cb[(size_t)idx[r] * DIM + c];
    float xi = x[e];
    float d  = q - xi;
    out[e] = xi + d;                            // inputs + (quantized - inputs)
    sred[threadIdx.x] = d * d;
    __syncthreads();
    for (int s = 128; s > 0; s >>= 1) {
        if (threadIdx.x < s) sred[threadIdx.x] += sred[threadIdx.x + s];
        __syncthreads();
    }
    if (threadIdx.x == 0) partials[blockIdx.x] = sred[0];
}

__global__ void k_final(const float* __restrict__ partials, float* __restrict__ loss_out) {
    __shared__ float sred[256];
    float s = 0.f;
    for (int i = threadIdx.x; i < 65536; i += 256) s += partials[i];  // fixed order
    sred[threadIdx.x] = s;
    __syncthreads();
    for (int st = 128; st > 0; st >>= 1) {
        if (threadIdx.x < st) sred[threadIdx.x] += sred[threadIdx.x + st];
        __syncthreads();
    }
    if (threadIdx.x == 0)
        loss_out[0] = 1.25f * (sred[0] / 16777216.0f);  // q_loss + 0.25*e_loss
}

extern "C" void kernel_launch(void* const* d_in, const int* in_sizes, int n_in,
                              void* d_out, int out_size, void* d_ws, size_t ws_size,
                              hipStream_t stream) {
    (void)in_sizes; (void)n_in; (void)out_size; (void)ws_size;
    const float* x  = (const float*)d_in[0];   // inputs [64,256,32,32]
    const float* cb = (const float*)d_in[1];   // codebook [1024,256]
    float* out = (float*)d_out;

    char* ws = (char*)d_ws;
    __bf16* bswz     = (__bf16*)ws;                              // 1,048,576 B
    float*  cnorm    = (float*)(ws + 1048576);                   //     4,096 B
    int*    idx      = (int*)  (ws + 1048576 + 4096);            //   262,144 B
    float*  partials = (float*)(ws + 1048576 + 4096 + 262144);   //   262,144 B

    k_cnorm <<<4,     256, 0, stream>>>(cb, cnorm);
    k_bswz  <<<64,    256, 0, stream>>>(cb, bswz);
    k_main  <<<512,   256, 0, stream>>>(x, bswz, cnorm, idx);
    k_gather<<<65536, 256, 0, stream>>>(x, cb, idx, out, partials);
    k_final <<<1,     256, 0, stream>>>(partials, out + TOT_ELEMS);
}
